// SS2D_72533407695500
// MI455X (gfx1250) — matrix operand order repaired and checked by reading, not compiled, not run on hardware
//
#include <hip/hip_runtime.h>

#define DEV __device__ __forceinline__

typedef __attribute__((ext_vector_type(16))) _Float16 v16h;
typedef __attribute__((ext_vector_type(8)))  float    v8f;
typedef __attribute__((ext_vector_type(4)))  unsigned v4u;
typedef __attribute__((ext_vector_type(8)))  int      v8i;
typedef __attribute__((ext_vector_type(4)))  int      v4i;

constexpr int B_  = 2;
constexpr int L_  = 4096;   // 64*64
constexpr int DM  = 96;     // d_model
constexpr int DI  = 192;    // d_inner
constexpr int NS  = 16;     // d_state
constexpr int RR  = 6;      // dt_rank
constexpr int KK  = 4;      // directions
constexpr int BL  = B_ * L_;  // 8192
constexpr int XC  = 48;     // padded x_dbl cols (6 dt + 16 B + 16 C = 38 -> 48)
constexpr int CH  = 64;     // scan: rows per LDS-staged chunk (64*48*4B = 12 KB)
constexpr int NCH = L_ / CH;

// ---- workspace layout (float offsets) --------------------------------------
constexpr long Z_OFF     = 0;                              // (B,L,DI)
constexpr long XIPRE_OFF = Z_OFF     + (long)BL*DI;        // (B,DI,L)
constexpr long XI_OFF    = XIPRE_OFF + (long)BL*DI;        // (B,DI,L)
constexpr long TACT_OFF  = XI_OFF    + (long)BL*DI;        // (B,L,DM)
constexpr long DTADD_OFF = TACT_OFF  + (long)BL*DM;        // (B,L,K*DI)
constexpr long XDBL_OFF  = DTADD_OFF + (long)BL*KK*DI;     // (B,K,L,48)
constexpr long OUTY_OFF  = XDBL_OFF  + (long)B_*KK*L_*XC;  // (B,K,DI,L)
constexpr long YN_OFF    = OUTY_OFF  + (long)B_*KK*DI*L_;  // (B,L,DI)
constexpr long XIT_OFF   = YN_OFF    + (long)BL*DI;        // (B,L,DI) token-major xi
// total = XIT_OFF + BL*DI = 22,806,528 floats (~91 MB)

DEV int a_kk(int v, int half) { return (v < 4 ? 2*v : 16 + 2*(v-4)) + 8*half; }
DEV int b_kk(int v, int half) { return (v < 4 ? 2*v : 16 + 2*(v-4)) + 8*half; }
DEV int tmap(int l) { return ((l & 63) << 6) | (l >> 6); }   // (h,w)<->(w,h), H=W=64
DEV float siluf(float x) { return x / (1.f + __expf(-x)); }

DEV v8f wmma16(v16h a, v16h b, v8f c) {
  return __builtin_amdgcn_wmma_f32_16x16x32_f16(false, a, false, b, (short)0, c,
                                                false, false);
}

#if __has_builtin(__builtin_amdgcn_tensor_load_to_lds) && \
    __has_builtin(__builtin_amdgcn_s_wait_tensorcnt)
#define SS2D_TDM 1
#endif

#if defined(SS2D_TDM)
// Issue one TDM 2D-tile DMA: 64 rows x 48 f32, contiguous (stride 48), into LDS.
// D# bitfields per CDNA5 ISA ch.8 (group0: count/lds/global/type, group1: dims).
// Toolchain probe: clang-23 builtin takes 6 args (g0 v4u, g1 v8i, g2 v4i,
// g3 v4i, g4 v8i, cpol); groups 2..4 unused for a 2D tensor -> zeros.
DEV void tdm_stage(const float* gsrc, unsigned lds_off) {
  unsigned long long ga = (unsigned long long)(const void*)gsrc;
  v4u g0;
  g0[0] = 1u;                                         // count=1, user descriptor
  g0[1] = lds_off;                                    // lds_addr (bytes)
  g0[2] = (unsigned)(ga & 0xFFFFFFFFull);             // global_addr[31:0]
  g0[3] = (unsigned)((ga >> 32) & 0x1FFFFFFull)       // global_addr[56:32]
        | (2u << 30);                                 // type=2 ("image")
  v8i g1;
  g1[0] = (int)(2u << 16);                            // data_size=2 (4 bytes)
  g1[1] = (int)((unsigned)XC << 16);                  // tensor_dim0[15:0] @ [63:48]
  g1[2] = (int)(((unsigned)L_ & 0xFFFFu) << 16);      // tensor_dim1[15:0] @ [95:80]
  g1[3] = (int)((unsigned)XC << 16);                  // tile_dim0=48 @ [127:112]
  g1[4] = CH;                                         // tile_dim1=64 @ [143:128]
  g1[5] = XC;                                         // tensor_dim0_stride=48
  g1[6] = 0;
  g1[7] = 0;
  v4i g2 = {0, 0, 0, 0};
  v4i g3 = {0, 0, 0, 0};
  v8i g4 = {0, 0, 0, 0, 0, 0, 0, 0};
  __builtin_amdgcn_tensor_load_to_lds(g0, g1, g2, g3, g4, 0);
}
#endif

// ---- K1: xz = x(8192x96) @ W_in(96x384); split -> xi_pre (B,DI,L), z (B,L,DI)
__global__ void k_gemm_xz(const float* __restrict__ x,
                          const float* __restrict__ Win,
                          float* __restrict__ xi_pre, float* __restrict__ z) {
  int wid  = (blockIdx.x * blockDim.x + threadIdx.x) >> 5;
  int lane = threadIdx.x & 31;
  int mtile = wid / 24, ntile = wid % 24;       // 512 x 24 tiles
  int lr = lane & 15, half = lane >> 4;
  v8f c = {};
  for (int kb = 0; kb < DM; kb += 32) {
    v16h a, b;
    const float* arow = x + (long)(mtile*16 + lr)*DM + kb;
#pragma unroll
    for (int v = 0; v < 8; ++v) {
      int ka = a_kk(v, half);
      a[2*v]   = (_Float16)arow[ka];
      a[2*v+1] = (_Float16)arow[ka+1];
      int kb2 = b_kk(v, half);
      b[2*v]   = (_Float16)Win[(long)(kb+kb2  )*384 + ntile*16 + lr];
      b[2*v+1] = (_Float16)Win[(long)(kb+kb2+1)*384 + ntile*16 + lr];
    }
    c = wmma16(a, b, c);
  }
  if (ntile < 12) {                              // uniform per wave: xi half
    int gcol = ntile*16 + lr;
#pragma unroll
    for (int v = 0; v < 8; ++v) {
      int grow = mtile*16 + v + 8*half;          // = b*L + l
      int bb = grow >> 12, l = grow & (L_-1);
      xi_pre[((long)bb*DI + gcol)*L_ + l] = c[v]; // channel-major for dwconv
    }
  } else {                                       // uniform per wave: z half
    int gcol = ntile*16 + lr - DI;
#pragma unroll
    for (int v = 0; v < 8; ++v) {
      int grow = mtile*16 + v + 8*half;
      z[(long)grow*DI + gcol] = c[v];
    }
  }
}

// ---- K2: xi = silu(dwconv3x3(xi_pre) + conv_b); write (B,DI,L) and (B,L,DI) -
__global__ void k_dwconv_xi(const float* __restrict__ xi_pre,
                            const float* __restrict__ cw,
                            const float* __restrict__ cb,
                            float* __restrict__ xi,
                            float* __restrict__ xi_t) {
  long idx = (long)blockIdx.x * blockDim.x + threadIdx.x;   // B*DI*L threads
  int l = idx % L_;  long rest = idx / L_;
  int cch = rest % DI; int b = rest / DI;
  int h = l >> 6, w = l & 63;
  const float* in = xi_pre + ((long)b*DI + cch)*L_;
  const float* wt = cw + cch*9;
  float acc = 0.f;
#pragma unroll
  for (int i = 0; i < 3; ++i) {
    int hh = h + i - 1; if (hh < 0 || hh > 63) continue;
#pragma unroll
    for (int j = 0; j < 3; ++j) {
      int ww2 = w + j - 1; if (ww2 < 0 || ww2 > 63) continue;
      acc += in[hh*64 + ww2] * wt[i*3 + j];
    }
  }
  float val = siluf(acc + cb[cch]);
  xi[idx] = val;                                  // (B,DI,L) for x_dbl gather
  xi_t[((long)b*L_ + l)*DI + cch] = val;          // (B,L,DI) for scan u-loads
}

// ---- K3: t_act = silu(dwconv3x3(d NHWC, prior_w1)), stored (B,L,DM) ---------
__global__ void k_dwconv_prior(const float* __restrict__ d,
                               const float* __restrict__ pw1,
                               float* __restrict__ t_act) {
  long idx = (long)blockIdx.x * blockDim.x + threadIdx.x;   // B*L*DM threads
  int cch = idx % DM; long rest = idx / DM;
  int l = rest % L_;  int b = rest / L_;
  int h = l >> 6, w = l & 63;
  const float* wt = pw1 + cch*9;
  float acc = 0.f;
#pragma unroll
  for (int i = 0; i < 3; ++i) {
    int hh = h + i - 1; if (hh < 0 || hh > 63) continue;
#pragma unroll
    for (int j = 0; j < 3; ++j) {
      int ww2 = w + j - 1; if (ww2 < 0 || ww2 > 63) continue;
      acc += d[(((long)b*64 + hh)*64 + ww2)*DM + cch] * wt[i*3 + j];  // NHWC in
    }
  }
  t_act[idx] = siluf(acc);
}

// ---- K4: dt_add = tanh(t_act(8192x96) @ prior_w2^T(96x768)) * alpha_dt ------
__global__ void k_gemm_prior(const float* __restrict__ t_act,
                             const float* __restrict__ pw2,
                             const float* __restrict__ alpha,
                             float* __restrict__ dt_add) {
  int wid  = (blockIdx.x * blockDim.x + threadIdx.x) >> 5;
  int lane = threadIdx.x & 31;
  int mtile = wid / 48, ntile = wid % 48;       // 512 x 48 tiles
  int lr = lane & 15, half = lane >> 4;
  v8f c = {};
  for (int kb = 0; kb < DM; kb += 32) {
    v16h a, b;
    const float* arow = t_act + (long)(mtile*16 + lr)*DM + kb;
    const float* brow = pw2 + (long)(ntile*16 + lr)*DM + kb;  // (o,c) row-major
#pragma unroll
    for (int v = 0; v < 8; ++v) {
      int ka = a_kk(v, half);
      a[2*v]   = (_Float16)arow[ka];
      a[2*v+1] = (_Float16)arow[ka+1];
      int kb2 = b_kk(v, half);
      b[2*v]   = (_Float16)brow[kb2];
      b[2*v+1] = (_Float16)brow[kb2+1];
    }
    c = wmma16(a, b, c);
  }
  int o = ntile*16 + lr;
  float al = alpha[o];
#pragma unroll
  for (int v = 0; v < 8; ++v) {
    int grow = mtile*16 + v + 8*half;
    dt_add[(long)grow*(KK*DI) + o] = tanhf(c[v]) * al;
  }
}

// ---- K5: x_dbl[b,k,l,0:38] = sum_d xs[b,k,d,l] * x_proj_w[k,c,d] (padded 48)
__global__ void k_gemm_xdbl(const float* __restrict__ xi,
                            const float* __restrict__ wp,
                            float* __restrict__ x_dbl) {
  int wid  = (blockIdx.x * blockDim.x + threadIdx.x) >> 5;  // 6144 tiles
  int lane = threadIdx.x & 31;
  int ntile = wid % 3;  int rest = wid / 3;
  int mtile = rest % 256; int bk = rest / 256;
  int b = bk >> 2, k = bk & 3;
  int lr = lane & 15, half = lane >> 4;
  int l = mtile*16 + lr;                         // scan index (A row)
  int p = (k == 0) ? l : (k == 1) ? tmap(l)
        : (k == 2) ? (L_-1-l) : tmap(L_-1-l);    // gather position in xi
  const float* xib = xi + (long)b*DI*L_;
  int bcol = ntile*16 + lr;
  v8f c = {};
  for (int kb = 0; kb < DI; kb += 32) {
    v16h a, bb;
#pragma unroll
    for (int v = 0; v < 8; ++v) {
      int ka = a_kk(v, half);
      a[2*v]   = (_Float16)xib[(long)(kb+ka  )*L_ + p];
      a[2*v+1] = (_Float16)xib[(long)(kb+ka+1)*L_ + p];
      int kb2 = b_kk(v, half);
      float w0 = (bcol < 38) ? wp[((long)k*38 + bcol)*DI + kb + kb2    ] : 0.f;
      float w1 = (bcol < 38) ? wp[((long)k*38 + bcol)*DI + kb + kb2 + 1] : 0.f;
      bb[2*v]   = (_Float16)w0;
      bb[2*v+1] = (_Float16)w1;
    }
    c = wmma16(a, bb, c);
  }
#pragma unroll
  for (int v = 0; v < 8; ++v) {
    int grow = mtile*16 + v + 8*half;
    x_dbl[((long)(b*KK + k)*L_ + grow)*XC + bcol] = c[v];
  }
}

// ---- K6: selective scan; one lane per (b,k,d) channel; x_dbl rows staged ----
// into double-buffered LDS chunks via Tensor Data Mover (TENSORcnt-tracked),
// DMA of chunk c+1 overlapped with the recurrence over chunk c.
__global__ void k_scan(const float* __restrict__ xi_t,
                       const float* __restrict__ x_dbl,
                       const float* __restrict__ dt_add,
                       const float* __restrict__ dtw,
                       const float* __restrict__ dtb,
                       const float* __restrict__ Alogs,
                       const float* __restrict__ Ds,
                       float* __restrict__ out_y) {
  __shared__ float sbuf[2][CH * XC];             // 2 x 12 KB of the 320 KB LDS
  int bk = blockIdx.x;  int b = bk >> 2, k = bk & 3;
  int d = threadIdx.x;                           // 0..191
  int cd = k*DI + d;
  float An[NS], h[NS], wr[RR];
#pragma unroll
  for (int n = 0; n < NS; ++n) { An[n] = -__expf(Alogs[(long)cd*NS + n]); h[n] = 0.f; }
#pragma unroll
  for (int r = 0; r < RR; ++r) wr[r] = dtw[(long)cd*RR + r];
  float bias = dtb[cd], Dd = Ds[cd];
  const float* xd  = x_dbl + (long)(b*KK + k)*L_*XC;
  const float* da  = dt_add + (long)b*L_*(KK*DI) + cd;
  const float* xtb = xi_t + (long)b*L_*DI + d;
  float* oy = out_y + ((long)(b*KK + k)*DI + d)*L_;

#if defined(SS2D_TDM)
  unsigned lbase = (unsigned)(unsigned long long)(void*)&sbuf[0][0];
  if (threadIdx.x < 32)                          // wave 0 drives the TDM
    tdm_stage(xd, lbase);                        // chunk 0
#else
  {
    const float4* src = (const float4*)xd;
    float4* dst = (float4*)&sbuf[0][0];
    for (int i = threadIdx.x; i < CH*XC/4; i += DI) dst[i] = src[i];
  }
#endif

  for (int c = 0; c < NCH; ++c) {
    if (c + 1 < NCH) {                           // prefetch next chunk
#if defined(SS2D_TDM)
      if (threadIdx.x < 32)
        tdm_stage(xd + (long)(c+1)*CH*XC,
                  lbase + (unsigned)(((c+1)&1)*CH*XC*4));
#else
      const float4* src = (const float4*)(xd + (long)(c+1)*CH*XC);
      float4* dst = (float4*)&sbuf[(c+1)&1][0];
      for (int i = threadIdx.x; i < CH*XC/4; i += DI) dst[i] = src[i];
#endif
    }
#if defined(SS2D_TDM)
    if (threadIdx.x < 32) {
      if (c + 1 < NCH) __builtin_amdgcn_s_wait_tensorcnt(1);  // chunk c done
      else             __builtin_amdgcn_s_wait_tensorcnt(0);
    }
#endif
    __syncthreads();                             // chunk c visible to all waves
    const float* chunk = &sbuf[c & 1][0];
    for (int ll = 0; ll < CH; ++ll) {
      int l = c*CH + ll;
      const float* xr = chunk + ll*XC;           // LDS broadcast across lanes
      float dts = bias + da[(long)l*(KK*DI)];
#pragma unroll
      for (int r = 0; r < RR; ++r) dts += xr[r] * wr[r];
      float delta = (dts > 20.f) ? dts : log1pf(__expf(dts));   // softplus
      int p = (k == 0) ? l : (k == 1) ? tmap(l)
            : (k == 2) ? (L_-1-l) : tmap(L_-1-l);
      float u = xtb[(long)p*DI];                 // coalesced: 192 lanes, 1 row
      float du = delta * u;
      float y = 0.f;
#pragma unroll
      for (int n = 0; n < NS; ++n) {
        float dA = __expf(delta * An[n]);
        h[n] = dA * h[n] + du * xr[RR + n];      // B cols 6..21
        y += h[n] * xr[RR + NS + n];             // C cols 22..37
      }
      oy[l] = y + u * Dd;
    }
    __syncthreads();                             // done reading before overwrite
  }
}

// ---- K7: un-map 4 directions, LayerNorm, SiLU(z) gate -> ynorm (B,L,DI) -----
__global__ void k_combine_ln(const float* __restrict__ out_y,
                             const float* __restrict__ z,
                             const float* __restrict__ gamma,
                             const float* __restrict__ beta,
                             float* __restrict__ ynorm) {
  __shared__ float sh[64 * DI];                 // 48 KB
  int idx = blockIdx.x * 64 + threadIdx.x;      // B*L threads total
  int b = idx >> 12, pos = idx & (L_-1);
  int tp = tmap(pos);
  const float* o0 = out_y + (long)(b*KK + 0)*DI*L_;
  const float* o1 = out_y + (long)(b*KK + 1)*DI*L_;
  const float* o2 = out_y + (long)(b*KK + 2)*DI*L_;
  const float* o3 = out_y + (long)(b*KK + 3)*DI*L_;
  float* my = sh + threadIdx.x * DI;
  float s = 0.f, s2 = 0.f;
  for (int d = 0; d < DI; ++d) {
    float y = o0[(long)d*L_ + pos]
            + o1[(long)d*L_ + tp]
            + o2[(long)d*L_ + (L_-1-pos)]
            + o3[(long)d*L_ + (L_-1-tp)];
    my[d] = y; s += y; s2 += y*y;
  }
  float mu = s * (1.f/DI);
  float var = s2 * (1.f/DI) - mu*mu;
  float rs = rsqrtf(var + 1e-5f);
  const float* zr = z + ((long)b*L_ + pos)*DI;
  float* yo = ynorm + ((long)b*L_ + pos)*DI;
  for (int d = 0; d < DI; ++d) {
    float g = siluf(zr[d]);
    yo[d] = ((my[d] - mu) * rs * gamma[d] + beta[d]) * g;
  }
}

// ---- K8: out = ynorm(8192x192) @ W_out(192x96) ------------------------------
__global__ void k_gemm_out(const float* __restrict__ yn,
                           const float* __restrict__ Wout,
                           float* __restrict__ out) {
  int wid  = (blockIdx.x * blockDim.x + threadIdx.x) >> 5;
  int lane = threadIdx.x & 31;
  int mtile = wid / 6, ntile = wid % 6;          // 512 x 6 tiles
  int lr = lane & 15, half = lane >> 4;
  v8f c = {};
  for (int kb = 0; kb < DI; kb += 32) {
    v16h a, b;
    const float* arow = yn + (long)(mtile*16 + lr)*DI + kb;
#pragma unroll
    for (int v = 0; v < 8; ++v) {
      int ka = a_kk(v, half);
      a[2*v]   = (_Float16)arow[ka];
      a[2*v+1] = (_Float16)arow[ka+1];
      int kb2 = b_kk(v, half);
      b[2*v]   = (_Float16)Wout[(long)(kb+kb2  )*DM + ntile*16 + lr];
      b[2*v+1] = (_Float16)Wout[(long)(kb+kb2+1)*DM + ntile*16 + lr];
    }
    c = wmma16(a, b, c);
  }
  int gcol = ntile*16 + lr;
#pragma unroll
  for (int v = 0; v < 8; ++v) {
    int grow = mtile*16 + v + 8*half;
    out[(long)grow*DM + gcol] = c[v];
  }
}

extern "C" void kernel_launch(void* const* d_in, const int* in_sizes, int n_in,
                              void* d_out, int out_size, void* d_ws, size_t ws_size,
                              hipStream_t stream) {
  const float* x     = (const float*)d_in[0];
  const float* dpr   = (const float*)d_in[1];
  const float* Win   = (const float*)d_in[2];
  const float* convw = (const float*)d_in[3];
  const float* convb = (const float*)d_in[4];
  const float* pw1   = (const float*)d_in[5];
  const float* pw2   = (const float*)d_in[6];
  const float* alpha = (const float*)d_in[7];
  const float* xpw   = (const float*)d_in[8];
  const float* dtw   = (const float*)d_in[9];
  const float* dtb   = (const float*)d_in[10];
  const float* Alogs = (const float*)d_in[11];
  const float* Ds    = (const float*)d_in[12];
  const float* lng   = (const float*)d_in[13];
  const float* lnb   = (const float*)d_in[14];
  const float* Wout  = (const float*)d_in[15];
  float* ws  = (float*)d_ws;
  float* out = (float*)d_out;

  float* z_buf   = ws + Z_OFF;
  float* xi_pre  = ws + XIPRE_OFF;
  float* xi      = ws + XI_OFF;
  float* t_act   = ws + TACT_OFF;
  float* dt_add  = ws + DTADD_OFF;
  float* x_dbl   = ws + XDBL_OFF;
  float* out_y   = ws + OUTY_OFF;
  float* ynorm   = ws + YN_OFF;
  float* xi_t    = ws + XIT_OFF;

  k_gemm_xz<<<1536, 256, 0, stream>>>(x, Win, xi_pre, z_buf);          // 512*24 tiles
  k_dwconv_xi<<<6144, 256, 0, stream>>>(xi_pre, convw, convb, xi, xi_t);
  k_dwconv_prior<<<3072, 256, 0, stream>>>(dpr, pw1, t_act);
  k_gemm_prior<<<3072, 256, 0, stream>>>(t_act, pw2, alpha, dt_add);   // 512*48 tiles
  k_gemm_xdbl<<<768, 256, 0, stream>>>(xi, xpw, x_dbl);                // 8*256*3 tiles
  k_scan<<<B_*KK, DI, 0, stream>>>(xi_t, x_dbl, dt_add, dtw, dtb, Alogs, Ds, out_y);
  k_combine_ln<<<BL/64, 64, 0, stream>>>(out_y, z_buf, lng, lnb, ynorm);
  k_gemm_out<<<384, 256, 0, stream>>>(ynorm, Wout, out);               // 512*6 tiles
}
